// QuadrupolePredictorSE3_57707180589679
// MI455X (gfx1250) — matrix operand order repaired and checked by reading, not compiled, not run on hardware
//
#include <hip/hip_runtime.h>
#include <hip/hip_bf16.h>
#include <math.h>

typedef __attribute__((ext_vector_type(16))) _Float16 v16h;
typedef __attribute__((ext_vector_type(8)))  _Float16 v8h;
typedef __attribute__((ext_vector_type(8)))  float    v8f;

#define NB 2
#define NN 256
#define HIDD 32

// workspace layout (float offsets)
#define OFS_Q0   0        // 16384 : q0 * 1/sqrt(8)
#define OFS_YK0  16384    // 16384 : x0n @ Mk[0]
#define OFS_YV0  32768    // 16384 : x0n @ Mv[0]
#define OFS_XA0  49152    // 2560  : x0 after attention residual
#define OFS_XF0  51712    // 2560  : x0 after ffn residual
#define OFS_BIAS 54272    // 64    : rb2 gathered for 4 col-tiles
#define OFS_ROWM 54336    // 4096 u32 : adjacency row bitmasks
#define OFS_ADJW 58432    // 4096 u32 : adjm (2-hop | eye) bitmasks
#define OFS_BPK  62528    // 2048 halves : R2 sub-cols in WMMA B-frag order

__device__ __forceinline__ float gelu_tanh(float x) {
  const float k0 = 0.7978845608028654f; // sqrt(2/pi)
  return 0.5f * x * (1.0f + tanhf(k0 * (x + 0.044715f * x * x * x)));
}

// ---- adjacency: row bitmasks from bool(uint8) adj ----
__global__ void k_rowmask(const unsigned char* __restrict__ adj, unsigned* __restrict__ rowm) {
  int t = blockIdx.x * blockDim.x + threadIdx.x;   // (b*256+i)*8+q
  if (t >= NB * NN * 8) return;
  int q = t & 7, bi = t >> 3;
  const unsigned char* row = adj + (size_t)bi * NN + q * 32;
  unsigned w = 0;
  #pragma unroll
  for (int k = 0; k < 32; k++) if (row[k]) w |= (1u << k);
  rowm[t] = w;
}

// ---- adjm = adj | (adj@adj > 0) | eye, as bitmask rows ----
__global__ void k_adjm(const unsigned* __restrict__ rowm, unsigned* __restrict__ adjw) {
  int t = blockIdx.x * blockDim.x + threadIdx.x;
  if (t >= NB * NN * 8) return;
  int q = t & 7, bi = t >> 3;
  int b = bi >> 8, i = bi & 255;
  unsigned res = rowm[bi * 8 + q];
  if ((i >> 5) == q) res |= (1u << (i & 31));
  for (int kw = 0; kw < 8; kw++) {
    unsigned m = rowm[bi * 8 + kw];
    while (m) {
      int k = kw * 32 + __ffs(m) - 1;
      m &= m - 1;
      res |= rowm[(b * NN + k) * 8 + q];
    }
  }
  adjw[t] = res;
}

// ---- norm_se3 (scalar path, x1==0) + tiny projections ----
__global__ void k_proj(const float* __restrict__ feats, const float* __restrict__ Wq,
                       const float* __restrict__ Mk, const float* __restrict__ Mv,
                       const float* __restrict__ gnA, const float* __restrict__ bnA,
                       float* __restrict__ q0s, float* __restrict__ yk0, float* __restrict__ yv0) {
  int t = blockIdx.x * blockDim.x + threadIdx.x;
  if (t >= NB * NN) return;
  float x[5], y0[5];
  #pragma unroll
  for (int c = 0; c < 5; c++) x[c] = feats[t * 5 + c];
  float mu = 0.f;
  #pragma unroll
  for (int c = 0; c < 5; c++) mu += fabsf(x[c]);
  mu *= 0.2f;
  float var = 0.f;
  #pragma unroll
  for (int c = 0; c < 5; c++) { float d = fabsf(x[c]) - mu; var += d * d; }
  var *= 0.2f;
  float sd = sqrtf(var) + 1e-8f;
  #pragma unroll
  for (int c = 0; c < 5; c++) {
    float v = (fabsf(x[c]) - mu) / sd * gnA[c] + bnA[c];
    float sg = (x[c] > 0.f) ? 1.f : ((x[c] < 0.f) ? -1.f : 0.f);
    y0[c] = sg * gelu_tanh(v);
  }
  for (int f = 0; f < HIDD; f++) {
    float a = 0.f, bk = 0.f, bv = 0.f;
    #pragma unroll
    for (int c = 0; c < 5; c++) {
      a  += y0[c] * Wq[c * HIDD + f];
      bk += y0[c] * Mk[c * HIDD + f];
      bv += y0[c] * Mv[c * HIDD + f];
    }
    q0s[t * HIDD + f] = a * 0.35355339059327373f;   // 1/sqrt(DH=8)
    yk0[t * HIDD + f] = bk;
    yv0[t * HIDD + f] = bv;
  }
}

// ---- pack needed R2 columns into WMMA B-fragment layout (f16) + rb2 bias ----
// col tiles: k00 -> cols 0..15, 16..31 ; v00 -> cols 192..207, 208..223
__global__ void k_pack(const float* __restrict__ R2, const float* __restrict__ rb2,
                       _Float16* __restrict__ Bpk, float* __restrict__ biasPk) {
  int tid = threadIdx.x;
  const int colbase[4] = {0, 16, 192, 208};
  if (tid < 128) {
    int ct = tid >> 5, L = tid & 31;
    int n = L & 15, g = L >> 4;
    #pragma unroll
    for (int t = 0; t < 16; t++) {
      int K = (t < 8 ? t : t + 8) + 8 * g;      // B frag: lanes 0-15 K=0..7,16..23, lanes 16-31 K=8..15,24..31
      int col = colbase[ct] + n;
      Bpk[(ct * 32 + L) * 16 + t] = (_Float16)R2[K * 384 + col];
    }
  }
  if (tid < 64) biasPk[tid] = rb2[colbase[tid >> 4] + (tid & 15)];
}

// ---- fused attention: one wave32 per (b,i) ----
__global__ __launch_bounds__(32) void k_attn(
    const float* __restrict__ feats, const float* __restrict__ coors,
    const float* __restrict__ R1, const float* __restrict__ rb1,
    const float* __restrict__ Wo0,
    const float* __restrict__ q0s, const float* __restrict__ yk0, const float* __restrict__ yv0,
    const _Float16* __restrict__ Bpk, const float* __restrict__ biasPk,
    const unsigned* __restrict__ adjw, float* __restrict__ xA0) {
  const int i = blockIdx.x, b = blockIdx.y;
  const int L = threadIdx.x, n16 = L & 15, g = L >> 4;

  __shared__ float l_att[4][NN];                     // logits -> attn
  __shared__ float l_d[NN];                          // distances to all j
  __shared__ float l_R1[32], l_rb1[32];
  __shared__ float l_o0[32];
  __shared__ __align__(16) _Float16 l_ke[16 * HIDD]; // current tile k-edge (f16)

  l_R1[L] = R1[L];
  l_rb1[L] = rb1[L];

  unsigned aw[8];
  #pragma unroll
  for (int q = 0; q < 8; q++) aw[q] = adjw[(b * NN + i) * 8 + q];

  float cix = coors[(b * NN + i) * 3 + 0];
  float ciy = coors[(b * NN + i) * 3 + 1];
  float ciz = coors[(b * NN + i) * 3 + 2];
  for (int j = L; j < NN; j += 32) {
    float rx = coors[(b * NN + j) * 3 + 0] - cix;
    float ry = coors[(b * NN + j) * 3 + 1] - ciy;
    float rz = coors[(b * NN + j) * 3 + 2] - ciz;
    l_d[j] = sqrtf(rx * rx + ry * ry + rz * rz + 1e-12f);
  }

  // B-fragment holding q as a block-diagonal head matrix: Bq[K=f][n=h] = q[f]*(h == f>>3)
  const float* q0row = q0s + (b * NN + i) * HIDD;
  v16h bq;
  #pragma unroll
  for (int t = 0; t < 16; t++) {
    int K = (t < 8 ? t : t + 8) + 8 * g;
    float qv = q0row[K];
    bq[t] = (n16 == (K >> 3)) ? (_Float16)qv : (_Float16)0.f;
  }

  v16h bf[4];
  float bias[4];
  #pragma unroll
  for (int ct = 0; ct < 4; ct++) {
    bf[ct] = *(const v16h*)(Bpk + (ct * 32 + L) * 16);
    bias[ct] = biasPk[ct * 16 + n16];
  }
  __syncthreads();

  // ---- pass 1: logits. phi via WMMA, then head-reduction via a second WMMA ----
  for (int jt = 0; jt < 16; jt++) {
    int jb = jt * 16;
    float dm = l_d[jb + n16];                 // A row M = n16
    v16h af;
    #pragma unroll
    for (int e = 0; e < 16; e++) {
      int K = (e < 8 ? e : e + 8) + 8 * g;    // ISA 16-bit A 16x32 layout
      float h = fmaf(dm, l_R1[K], l_rb1[K]);
      af[e] = (_Float16)fmaxf(h, 0.f);
    }
    #pragma unroll
    for (int ct = 0; ct < 2; ct++) {
      v8f cf;
      #pragma unroll
      for (int r = 0; r < 8; r++) cf[r] = bias[ct];
      v8f D = __builtin_amdgcn_wmma_f32_16x16x32_f16(false, af, false, bf[ct],
                                                     (short)0, cf, false, false);
      int f = ct * 16 + n16;
      #pragma unroll
      for (int r = 0; r < 8; r++) {
        int j = jb + r + 8 * g;               // D row M = r + 8g
        float ke = D[r] * yk0[(b * NN + j) * HIDD + f];
        l_ke[(r + 8 * g) * HIDD + f] = (_Float16)ke;   // [jrel][f]
      }
    }
    __syncthreads();
    // reload as A-fragment (row m = n16, K = f) : two contiguous 16B groups
    const _Float16* arow = l_ke + n16 * HIDD;
    v8h lo = *(const v8h*)(arow + 8 * g);
    v8h hi = *(const v8h*)(arow + 16 + 8 * g);
    v16h a2 = __builtin_shufflevector(lo, hi, 0, 1, 2, 3, 4, 5, 6, 7,
                                      8, 9, 10, 11, 12, 13, 14, 15);
    v8f c2;
    #pragma unroll
    for (int r = 0; r < 8; r++) c2[r] = 0.f;
    v8f D2 = __builtin_amdgcn_wmma_f32_16x16x32_f16(false, a2, false, bq,
                                                    (short)0, c2, false, false);
    if (n16 < 4) {                            // column n = head, rows = jrel
      #pragma unroll
      for (int r = 0; r < 8; r++) l_att[n16][jb + r + 8 * g] = D2[r];
    }
    __syncthreads();
  }

  // ---- masked softmax over j, per head (8 lanes per head) ----
  {
    int hh = L >> 3, sub = L & 7;
    float mx = -3e38f;
    for (int j = sub; j < NN; j += 8) {
      bool on = (aw[j >> 5] >> (j & 31)) & 1u;
      float lg = on ? l_att[hh][j] : -1e9f;
      mx = fmaxf(mx, lg);
    }
    mx = fmaxf(mx, __shfl_xor(mx, 1, 32));
    mx = fmaxf(mx, __shfl_xor(mx, 2, 32));
    mx = fmaxf(mx, __shfl_xor(mx, 4, 32));
    float sm = 0.f;
    for (int j = sub; j < NN; j += 8) {
      bool on = (aw[j >> 5] >> (j & 31)) & 1u;
      float lg = on ? l_att[hh][j] : -1e9f;
      sm += expf(lg - mx);
    }
    sm += __shfl_xor(sm, 1, 32);
    sm += __shfl_xor(sm, 2, 32);
    sm += __shfl_xor(sm, 4, 32);
    float inv = 1.0f / sm;
    for (int j = sub; j < NN; j += 8) {
      bool on = (aw[j >> 5] >> (j & 31)) & 1u;
      float lg = on ? l_att[hh][j] : -1e9f;
      l_att[hh][j] = expf(lg - mx) * inv;
    }
  }
  __syncthreads();

  // ---- pass 2: o0 = sum_j attn * (phi v00 * y00v) ----
  float a0[2] = {0.f, 0.f};
  for (int jt = 0; jt < 16; jt++) {
    int jb = jt * 16;
    float dm = l_d[jb + n16];
    v16h af;
    #pragma unroll
    for (int e = 0; e < 16; e++) {
      int K = (e < 8 ? e : e + 8) + 8 * g;
      float h = fmaf(dm, l_R1[K], l_rb1[K]);
      af[e] = (_Float16)fmaxf(h, 0.f);
    }
    #pragma unroll
    for (int ct = 2; ct < 4; ct++) {
      v8f cf;
      #pragma unroll
      for (int r = 0; r < 8; r++) cf[r] = bias[ct];
      v8f D = __builtin_amdgcn_wmma_f32_16x16x32_f16(false, af, false, bf[ct],
                                                     (short)0, cf, false, false);
      int fv = (ct - 2) * 16 + n16;
      int hh = fv >> 3;
      #pragma unroll
      for (int r = 0; r < 8; r++) {
        int j = jb + r + 8 * g;
        a0[ct - 2] += l_att[hh][j] * yv0[(b * NN + j) * HIDD + fv] * D[r];
      }
    }
  }
  a0[0] += __shfl_xor(a0[0], 16, 32);
  a0[1] += __shfl_xor(a0[1], 16, 32);
  if (g == 0) { l_o0[n16] = a0[0]; l_o0[16 + n16] = a0[1]; }
  __syncthreads();

  if (L < 5) {
    float s = feats[(b * NN + i) * 5 + L];            // residual
    #pragma unroll
    for (int hd = 0; hd < HIDD; hd++) s += l_o0[hd] * Wo0[hd * 5 + L];
    xA0[(b * NN + i) * 5 + L] = s;
  }
}

// ---- norm_se3 #2 (scalar path) + scalar feedforward + residual ----
__global__ void k_ffn(const float* __restrict__ xA0, const float* __restrict__ gnF,
                      const float* __restrict__ bnF, const float* __restrict__ Wf1,
                      const float* __restrict__ Wf2, float* __restrict__ xF0) {
  int t = blockIdx.x * blockDim.x + threadIdx.x;
  if (t >= NB * NN) return;
  float x[5];
  #pragma unroll
  for (int c = 0; c < 5; c++) x[c] = xA0[t * 5 + c];
  float mu = 0.f;
  #pragma unroll
  for (int c = 0; c < 5; c++) mu += fabsf(x[c]);
  mu *= 0.2f;
  float var = 0.f;
  #pragma unroll
  for (int c = 0; c < 5; c++) { float d = fabsf(x[c]) - mu; var += d * d; }
  var *= 0.2f;
  float sd = sqrtf(var) + 1e-8f;
  float y[5];
  #pragma unroll
  for (int c = 0; c < 5; c++) {
    float v = (fabsf(x[c]) - mu) / sd * gnF[c] + bnF[c];
    float sg = (x[c] > 0.f) ? 1.f : ((x[c] < 0.f) ? -1.f : 0.f);
    y[c] = sg * gelu_tanh(v);
  }
  float f0[5] = {0.f, 0.f, 0.f, 0.f, 0.f};
  for (int f = 0; f < 20; f++) {
    float h = 0.f;
    #pragma unroll
    for (int c = 0; c < 5; c++) h += y[c] * Wf1[c * 20 + f];
    h = gelu_tanh(h);
    #pragma unroll
    for (int c = 0; c < 5; c++) f0[c] += h * Wf2[f * 5 + c];
  }
  #pragma unroll
  for (int c = 0; c < 5; c++) xF0[t * 5 + c] = x[c] + f0[c];
}

// ---- mean-pool over nodes + 5->128->9 head ----
__global__ __launch_bounds__(256) void k_head(const float* __restrict__ xF0,
                                              const float* __restrict__ Wh1,
                                              const float* __restrict__ bh1,
                                              const float* __restrict__ Wh2,
                                              const float* __restrict__ bh2,
                                              float* __restrict__ out) {
  int b = blockIdx.x, tid = threadIdx.x;
  __shared__ float sm[256];
  __shared__ float pool[5];
  __shared__ float hbuf[128];
  float px[5];
  #pragma unroll
  for (int c = 0; c < 5; c++) px[c] = xF0[(b * NN + tid) * 5 + c];
  for (int c = 0; c < 5; c++) {
    sm[tid] = px[c];
    __syncthreads();
    for (int s = 128; s > 0; s >>= 1) {
      if (tid < s) sm[tid] += sm[tid + s];
      __syncthreads();
    }
    if (tid == 0) pool[c] = sm[0] * (1.0f / NN);
    __syncthreads();
  }
  if (tid < 128) {
    float h = bh1[tid];
    #pragma unroll
    for (int c = 0; c < 5; c++) h += pool[c] * Wh1[c * 128 + tid];
    hbuf[tid] = fmaxf(h, 0.f);
  }
  __syncthreads();
  if (tid < 9) {
    float s = bh2[tid];
    for (int k = 0; k < 128; k++) s += hbuf[k] * Wh2[k * 9 + tid];
    out[b * 9 + tid] = s;
  }
}

extern "C" void kernel_launch(void* const* d_in, const int* in_sizes, int n_in,
                              void* d_out, int out_size, void* d_ws, size_t ws_size,
                              hipStream_t stream) {
  (void)in_sizes; (void)n_in; (void)out_size; (void)ws_size;
  const float* feats = (const float*)d_in[0];
  const float* coors = (const float*)d_in[1];
  const unsigned char* adj = (const unsigned char*)d_in[2];
  const float* Wq  = (const float*)d_in[3];
  const float* Mk  = (const float*)d_in[4];
  const float* Mv  = (const float*)d_in[5];
  const float* R1  = (const float*)d_in[6];
  const float* rb1 = (const float*)d_in[7];
  const float* R2  = (const float*)d_in[8];
  const float* rb2 = (const float*)d_in[9];
  const float* Wo0 = (const float*)d_in[10];
  const float* gnA = (const float*)d_in[12];
  const float* bnA = (const float*)d_in[13];
  const float* gnF = (const float*)d_in[14];
  const float* bnF = (const float*)d_in[15];
  const float* Wf1 = (const float*)d_in[16];
  const float* Wf2 = (const float*)d_in[17];
  const float* Wh1 = (const float*)d_in[20];
  const float* bh1 = (const float*)d_in[21];
  const float* Wh2 = (const float*)d_in[22];
  const float* bh2 = (const float*)d_in[23];
  float* out = (float*)d_out;

  float* ws = (float*)d_ws;
  float* q0s = ws + OFS_Q0;
  float* yk0 = ws + OFS_YK0;
  float* yv0 = ws + OFS_YV0;
  float* xA0 = ws + OFS_XA0;
  float* xF0 = ws + OFS_XF0;
  float* biasPk = ws + OFS_BIAS;
  unsigned* rowm = (unsigned*)(ws + OFS_ROWM);
  unsigned* adjw = (unsigned*)(ws + OFS_ADJW);
  _Float16* Bpk = (_Float16*)(ws + OFS_BPK);

  k_rowmask<<<16, 256, 0, stream>>>(adj, rowm);
  k_adjm<<<16, 256, 0, stream>>>(rowm, adjw);
  k_proj<<<2, 256, 0, stream>>>(feats, Wq, Mk, Mv, gnA, bnA, q0s, yk0, yv0);
  k_pack<<<1, 128, 0, stream>>>(R2, rb2, Bpk, biasPk);
  k_attn<<<dim3(NN, NB), 32, 0, stream>>>(feats, coors, R1, rb1, Wo0,
                                          q0s, yk0, yv0, Bpk, biasPk, adjw, xA0);
  k_ffn<<<2, 256, 0, stream>>>(xA0, gnF, bnF, Wf1, Wf2, xF0);
  k_head<<<NB, 256, 0, stream>>>(xF0, Wh1, bh1, Wh2, bh2, out);
}
